// ConvLocalAttention_44281112821891
// MI455X (gfx1250) — matrix operand order repaired and checked by reading, not compiled, not run on hardware
//
#include <hip/hip_runtime.h>
#include <hip/hip_bf16.h>

typedef __attribute__((ext_vector_type(16))) _Float16 v16h;
typedef __attribute__((ext_vector_type(8)))  _Float16 h8;
typedef __attribute__((ext_vector_type(4)))  _Float16 h4;
typedef __attribute__((ext_vector_type(8)))  float    v8f;
typedef __attribute__((ext_vector_type(4)))  float    f32x4;

#define HEADS 8
#define DHEAD 64
#define NSEQ  2048
#define DIMIN 512
#define INNER 512           // HEADS*DHEAD
#define OQKV  1536          // 3*INNER
#define BATCH 8
#define NEG_INF (-3.4028234663852886e38f)

#define SHUF16(A, B) __builtin_shufflevector((A), (B), 0,1,2,3,4,5,6,7,8,9,10,11,12,13,14,15)

__device__ __forceinline__ h4 load4_f16(const float* p) {
    f32x4 v = *(const f32x4*)p;
    return (h4){(_Float16)v[0], (_Float16)v[1], (_Float16)v[2], (_Float16)v[3]};
}
__device__ __forceinline__ h4 load4_f16(const _Float16* p) {
    return *(const h4*)p;
}

// ---------------------------------------------------------------------------
// Generic WMMA GEMM:  Out[b][o][n] = sum_k W[o][k] * X[b][k][n]
// Block 256 threads (8 waves); tile 64(M) x 64(N); K stepped by 32.
// Wave grid 4(M) x 2(N); each wave: one A-frag, two B-frags, two WMMAs/step.
// All fragments are contiguous 16B-aligned LDS runs -> ds_load_b128.
// ---------------------------------------------------------------------------
template <typename TB, typename TOut>
__global__ __launch_bounds__(256)
void gemm_wmma(const float* __restrict__ W, const TB* __restrict__ X,
               TOut* __restrict__ Out, int O, int K, int N) {
    const int bidx = blockIdx.z;
    const int o0   = blockIdx.y * 64;
    const int n0   = blockIdx.x * 64;

    __shared__ _Float16 As [64][40];   // [m][k]   row stride 80B (16B mult)
    __shared__ _Float16 BsT[64][40];   // [n][k]   transposed B tile

    const int tid  = threadIdx.x;
    const int wid  = tid >> 5;
    const int lane = tid & 31;
    const int l    = lane & 15;
    const int hi   = lane >> 4;
    const int lo   = hi * 8;
    const int klo  = hi * 16;
    const int mb   = 16 * (wid >> 1);
    const int nb   = 32 * (wid & 1);

    v8f c0 = {0.f,0.f,0.f,0.f,0.f,0.f,0.f,0.f};
    v8f c1 = {0.f,0.f,0.f,0.f,0.f,0.f,0.f,0.f};

    for (int k0 = 0; k0 < K; k0 += 32) {
        __syncthreads();
        // ---- A tile: 64x32 f32, float4 loads -> h4 LDS stores ----
        #pragma unroll
        for (int u = 0; u < 2; ++u) {
            int idx = tid + u * 256;             // 0..511
            int mo = idx >> 3, kg = idx & 7;     // kg: 8 groups of 4 k
            h4 hv = load4_f16(&W[(size_t)(o0 + mo) * K + k0 + kg * 4]);
            *(h4*)&As[mo][kg * 4] = hv;
        }
        // ---- B tile: 32x64, vector global loads, transposed LDS stores ----
        #pragma unroll
        for (int u = 0; u < 2; ++u) {
            int idx = tid + u * 256;             // 0..511
            int kk = idx >> 4, ng = idx & 15;    // ng: 16 groups of 4 n
            h4 hv = load4_f16(&X[((size_t)bidx * K + k0 + kk) * N + n0 + ng * 4]);
            BsT[ng * 4 + 0][kk] = hv[0];
            BsT[ng * 4 + 1][kk] = hv[1];
            BsT[ng * 4 + 2][kk] = hv[2];
            BsT[ng * 4 + 3][kk] = hv[3];
        }
        // prefetch next K slab (global_prefetch_b8 path)
        if (k0 + 32 < K) {
            __builtin_prefetch(&W[(size_t)(o0 + (tid & 63)) * K + k0 + 32], 0, 1);
            __builtin_prefetch(&X[((size_t)bidx * K + k0 + 32 + (tid & 31)) * N + n0], 0, 1);
        }
        __syncthreads();

        h8 a_lo = *(const h8*)&As[mb + l][klo];       // K = klo..klo+7
        h8 a_hi = *(const h8*)&As[mb + l][klo + 8];   // K = klo+8..klo+15
        v16h a  = SHUF16(a_lo, a_hi);

        h8 b0_lo = *(const h8*)&BsT[nb + l][klo];     // K = klo..klo+15
        h8 b0_hi = *(const h8*)&BsT[nb + l][klo + 8];
        v16h b0  = SHUF16(b0_lo, b0_hi);
        h8 b1_lo = *(const h8*)&BsT[nb + 16 + l][klo];
        h8 b1_hi = *(const h8*)&BsT[nb + 16 + l][klo + 8];
        v16h b1  = SHUF16(b1_lo, b1_hi);

        c0 = __builtin_amdgcn_wmma_f32_16x16x32_f16(false, a, false, b0,
                                                    (short)0, c0, false, false);
        c1 = __builtin_amdgcn_wmma_f32_16x16x32_f16(false, a, false, b1,
                                                    (short)0, c1, false, false);
    }

    const int col0 = n0 + nb + l;
    const int col1 = col0 + 16;
    #pragma unroll
    for (int r = 0; r < 8; ++r) {
        int row = o0 + mb + r + hi * 8;
        Out[((size_t)bidx * O + row) * N + col0] = (TOut)c0[r];
        Out[((size_t)bidx * O + row) * N + col1] = (TOut)c1[r];
    }
}

// ---------------------------------------------------------------------------
// In-place L2 normalize q,k (layout qkv[b][o][n], o = qk*512 + h*64 + d).
// Folds per-dim scale and QK_SCALE=8 into q.
// ---------------------------------------------------------------------------
__global__ __launch_bounds__(256)
void normalize_qk(_Float16* __restrict__ qkv,
                  const float* __restrict__ q_scale,
                  const float* __restrict__ k_scale) {
    int gid = blockIdx.x * 256 + threadIdx.x;       // 2*8*8*2048 = 262144
    int qk  = gid >> 17;
    int rem = gid & 131071;
    int bh  = rem >> 11;
    int n   = rem & 2047;
    int b = bh >> 3, h = bh & 7;

    size_t base = ((size_t)(b * OQKV + qk * INNER + h * DHEAD)) * NSEQ + n;
    float vals[DHEAD];
    float ss = 0.f;
    #pragma unroll 8
    for (int d = 0; d < DHEAD; ++d) {
        float v = (float)qkv[base + (size_t)d * NSEQ];
        vals[d] = v;
        ss += v * v;
    }
    float inv = 1.0f / fmaxf(sqrtf(ss), 1e-12f);
    const float* sc = qk ? k_scale : q_scale;
    float extra = qk ? 1.0f : 8.0f;                 // QK_SCALE folded into q
    #pragma unroll 8
    for (int d = 0; d < DHEAD; ++d)
        qkv[base + (size_t)d * NSEQ] = (_Float16)(vals[d] * inv * sc[d] * extra);
}

// ---------------------------------------------------------------------------
// Local attention. Block = (b, h, window, half): 128 threads = 4 waves,
// 64 queries; keys span 192 positions of [prev-window || cur-window].
// K row-major [j][d] (sim B-frags contiguous); V d-major [d][j] (PV B-frags
// and the global->LDS copy both contiguous).
// ---------------------------------------------------------------------------
__global__ __launch_bounds__(128)
void attn_kernel(const _Float16* __restrict__ qkv, _Float16* __restrict__ att) {
    const int hh  = blockIdx.x;          // half of window: 0/1
    const int win = blockIdx.y;          // 0..15
    const int bh  = blockIdx.z;          // 0..63
    const int b = bh >> 3, h = bh & 7;
    const int n0 = win * 128;
    const int Ibase = hh * 64;

    __shared__ _Float16 Ks [192][72];    // [j][d]  stride 144B
    __shared__ _Float16 VsT[64][200];    // [d][j]  stride 400B
    __shared__ _Float16 Pb [4][16][32];  // per-wave P chunk

    const int tid  = threadIdx.x;
    const int w    = tid >> 5;
    const int lane = tid & 31;
    const int l    = lane & 15;
    const int hi   = lane >> 4;
    const int lo   = hi * 8;
    const int klo  = hi * 16;
    const size_t bstride = (size_t)b * OQKV * NSEQ;

    // ---- load K, V tiles: h4 global loads; V copies contiguous ----
    #pragma unroll 4
    for (int u = 0; u < 24; ++u) {
        int idx = tid + u * 128;         // 0..3071 groups of 4 j
        int d = idx / 48;
        int j0 = (idx % 48) * 4;
        int tk0 = n0 - 128 + Ibase + j0;
        h4 kk4 = {(_Float16)0.f, (_Float16)0.f, (_Float16)0.f, (_Float16)0.f};
        h4 vv4 = kk4;
        if (tk0 >= 0) {                  // group-aligned boundary (tk0 mult of 4)
            kk4 = *(const h4*)&qkv[bstride + (size_t)(INNER     + h * DHEAD + d) * NSEQ + tk0];
            vv4 = *(const h4*)&qkv[bstride + (size_t)(2 * INNER + h * DHEAD + d) * NSEQ + tk0];
        }
        *(h4*)&VsT[d][j0] = vv4;
        Ks[j0 + 0][d] = kk4[0];
        Ks[j0 + 1][d] = kk4[1];
        Ks[j0 + 2][d] = kk4[2];
        Ks[j0 + 3][d] = kk4[3];
    }
    __syncthreads();

    // ---- Q fragments from global (each wave owns its 16 rows) ----
    const int nq = n0 + Ibase + 16 * w + l;          // A-frag row position
    v16h aq0, aq1;
    #pragma unroll
    for (int i = 0; i < 16; ++i) {
        int d0 = klo + i;
        aq0[i] = qkv[bstride + (size_t)(h * DHEAD + d0)      * NSEQ + nq];
        aq1[i] = qkv[bstride + (size_t)(h * DHEAD + 32 + d0) * NSEQ + nq];
    }

    // ---- sim = Q * K^T : 12 key-tiles of 16, d in two 32-chunks ----
    v8f cs[12];
    #pragma unroll
    for (int t = 0; t < 12; ++t) cs[t] = (v8f){0.f,0.f,0.f,0.f,0.f,0.f,0.f,0.f};
    #pragma unroll
    for (int t = 0; t < 12; ++t) {
        const _Float16* krow = &Ks[16 * t + l][0];
        v16h bk0 = SHUF16(*(const h8*)&krow[klo],      *(const h8*)&krow[klo + 8]);
        v16h bk1 = SHUF16(*(const h8*)&krow[32 + klo], *(const h8*)&krow[40 + klo]);
        cs[t] = __builtin_amdgcn_wmma_f32_16x16x32_f16(false, aq0, false, bk0,
                                                       (short)0, cs[t], false, false);
        cs[t] = __builtin_amdgcn_wmma_f32_16x16x32_f16(false, aq1, false, bk1,
                                                       (short)0, cs[t], false, false);
    }

    // ---- mask + softmax (C-layout: row = r + 8*hi, col = 16t + l) ----
    #pragma unroll
    for (int r = 0; r < 8; ++r) {
        int i = Ibase + 16 * w + r + hi * 8;         // window-local query index
        float mx = NEG_INF;
        #pragma unroll
        for (int t = 0; t < 12; ++t) {
            int j = Ibase + 16 * t + l;              // concat-key index
            bool ok = (j >= i) && (j <= i + 128) && (win > 0 || j >= 128);
            float s = ok ? cs[t][r] : NEG_INF;
            cs[t][r] = s;
            mx = fmaxf(mx, s);
        }
        #pragma unroll
        for (int off = 1; off < 16; off <<= 1)
            mx = fmaxf(mx, __shfl_xor(mx, off, 32));
        float sum = 0.f;
        #pragma unroll
        for (int t = 0; t < 12; ++t) {
            float p = expf(cs[t][r] - mx);
            cs[t][r] = p;
            sum += p;
        }
        #pragma unroll
        for (int off = 1; off < 16; off <<= 1)
            sum += __shfl_xor(sum, off, 32);
        float inv = 1.0f / sum;
        #pragma unroll
        for (int t = 0; t < 12; ++t) cs[t][r] *= inv;
    }

    // ---- O = P * V : 6 key-chunks of 32, 4 d-tiles of 16 ----
    v8f oc[4];
    #pragma unroll
    for (int dt = 0; dt < 4; ++dt) oc[dt] = (v8f){0.f,0.f,0.f,0.f,0.f,0.f,0.f,0.f};
    #pragma unroll
    for (int kc = 0; kc < 6; ++kc) {
        __syncthreads();
        #pragma unroll
        for (int tt = 0; tt < 2; ++tt) {             // stash P chunk (C -> A reshape)
            int t = 2 * kc + tt;
            #pragma unroll
            for (int r = 0; r < 8; ++r)
                Pb[w][r + hi * 8][tt * 16 + l] = (_Float16)cs[t][r];
        }
        __syncthreads();
        v16h ap = SHUF16(*(const h8*)&Pb[w][l][klo], *(const h8*)&Pb[w][l][klo + 8]);
        #pragma unroll
        for (int dt = 0; dt < 4; ++dt) {
            const _Float16* vrow = &VsT[16 * dt + l][32 * kc];
            v16h bv = SHUF16(*(const h8*)&vrow[klo], *(const h8*)&vrow[klo + 8]);
            oc[dt] = __builtin_amdgcn_wmma_f32_16x16x32_f16(false, ap, false, bv,
                                                            (short)0, oc[dt], false, false);
        }
    }

    // ---- store: 8 C-rows are consecutive n positions -> one 16B store ----
    const int iq0 = n0 + Ibase + 16 * w + hi * 8;
    #pragma unroll
    for (int dt = 0; dt < 4; ++dt) {
        h8 ov;
        #pragma unroll
        for (int r = 0; r < 8; ++r) ov[r] = (_Float16)oc[dt][r];
        *(h8*)&att[((size_t)b * INNER + h * DHEAD + 16 * dt + l) * NSEQ + iq0] = ov;
    }
}

// ---------------------------------------------------------------------------
extern "C" void kernel_launch(void* const* d_in, const int* in_sizes, int n_in,
                              void* d_out, int out_size, void* d_ws, size_t ws_size,
                              hipStream_t stream) {
    const float* x       = (const float*)d_in[0];  // [8][512][2048]
    const float* w_qkv   = (const float*)d_in[1];  // [1536][512]
    const float* w_out   = (const float*)d_in[2];  // [512][512]
    const float* q_scale = (const float*)d_in[3];  // [64]
    const float* k_scale = (const float*)d_in[4];  // [64]
    float* out = (float*)d_out;                    // [8][512][2048]

    _Float16* qkv = (_Float16*)d_ws;                                            // 50.3 MB
    _Float16* att = (_Float16*)((char*)d_ws + (size_t)BATCH * OQKV * NSEQ * 2); // +16.8 MB

    // 1) QKV projection (f32 in -> f16 out)
    gemm_wmma<float, _Float16><<<dim3(NSEQ / 64, OQKV / 64, BATCH), 256, 0, stream>>>(
        w_qkv, x, qkv, OQKV, DIMIN, NSEQ);

    // 2) in-place q/k normalization (+ scales + QK_SCALE)
    normalize_qk<<<dim3((2 * BATCH * HEADS * NSEQ) / 256), 256, 0, stream>>>(
        qkv, q_scale, k_scale);

    // 3) windowed attention
    attn_kernel<<<dim3(2, NSEQ / 128, BATCH * HEADS), 128, 0, stream>>>(qkv, att);

    // 4) output projection (f16 in -> f32 out)
    gemm_wmma<_Float16, float><<<dim3(NSEQ / 64, INNER / 64, BATCH), 256, 0, stream>>>(
        w_out, att, out, INNER, INNER, NSEQ);
}
